// PCNClassifier_76519137345576
// MI455X (gfx1250) — matrix operand order repaired and checked by reading, not compiled, not run on hardware
//
#include <hip/hip_runtime.h>
#include <hip/hip_bf16.h>

typedef __attribute__((ext_vector_type(16))) _Float16 v16h;
typedef __attribute__((ext_vector_type(8)))  float    v8f;

#define BNI 0.999995000037f   /* 1/sqrt(1+1e-5) */
#define NEG_BIG (-3.0e38f)

enum { Bb = 16, Nn = 2048, Kk = 32, M1 = 1024, M2 = 256 };

// ---------------- wave-scope LDS fence (producer lanes -> consumer lanes) ----
__device__ __forceinline__ void lds_fence_wave() {
  __builtin_amdgcn_wave_barrier();
  asm volatile("s_wait_dscnt 0x0" ::: "memory");
  __builtin_amdgcn_wave_barrier();
}

// ---------------- WMMA fragment loaders ------------------------------------
// A: row-major [16 x 32] tile at (row0, kc) with leading dim ld.
// lanes 0-15 hold K = kc+0..7 , kc+16..23 ; lanes 16-31 hold K = kc+8..15, kc+24..31
__device__ __forceinline__ v16h frag_a(const _Float16* A, int ld, int row0, int kc, int lane) {
  const _Float16* p = A + (row0 + (lane & 15)) * ld + kc + ((lane & 16) ? 8 : 0);
  v16h a;
#pragma unroll
  for (int i = 0; i < 8; ++i) a[i] = p[i];
#pragma unroll
  for (int i = 0; i < 8; ++i) a[8 + i] = p[16 + i];
  return a;
}
// B: pre-swizzled weights. For tile (ct, kcIdx) lane L reads 16 contiguous halves.
__device__ __forceinline__ v16h frag_b(const _Float16* Wsw, int nd, int ct, int kcIdx, int lane) {
  const _Float16* p = Wsw + ((size_t)(kcIdx * (nd >> 4) + ct) * 32 + lane) * 16;
  v16h b;
#pragma unroll
  for (int i = 0; i < 16; ++i) b[i] = p[i];
  return b;
}

// One MLP layer on a 32-row tile (wave scope), A fragments cached in registers.
template <int KD, int ND>
__device__ __forceinline__ void wave_layer_t(const _Float16* A, const _Float16* Wsw,
                                             const float* sc, const float* sh, bool relu,
                                             _Float16* H, int lane) {
  constexpr int KCH = KD / 32;
#pragma unroll
  for (int rt = 0; rt < 2; ++rt) {
    v16h af[KCH];
#pragma unroll
    for (int kk = 0; kk < KCH; ++kk) af[kk] = frag_a(A, KD, rt * 16, kk * 32, lane);
    for (int ct = 0; ct < ND / 16; ++ct) {
      v8f c = {};
#pragma unroll
      for (int kk = 0; kk < KCH; ++kk) {
        v16h b = frag_b(Wsw, ND, ct, kk, lane);
        c = __builtin_amdgcn_wmma_f32_16x16x32_f16(false, af[kk], false, b, (short)0, c, false, false);
      }
      int col = ct * 16 + (lane & 15);
      float s = sc[col], h0 = sh[col];
      int roff = rt * 16 + ((lane & 16) ? 8 : 0);
#pragma unroll
      for (int j = 0; j < 8; ++j) {
        float v = fmaf(c[j], s, h0);
        if (relu) v = fmaxf(v, 0.f);
        H[(roff + j) * ND + col] = (_Float16)v;
      }
    }
  }
}

// Final layer + masked max-aggregation over the 32 neighbor rows.
template <int KD, int ND>
__device__ __forceinline__ void wave_layer_max_t(const _Float16* A, const _Float16* Wsw,
                                                 const float* sc, const float* sh,
                                                 unsigned vmask, _Float16* out, int lane) {
  constexpr int KCH = KD / 32;
  constexpr int NT = ND / 16;
  float mx[NT];
#pragma unroll
  for (int ct = 0; ct < NT; ++ct) mx[ct] = NEG_BIG;
#pragma unroll
  for (int rt = 0; rt < 2; ++rt) {
    v16h af[KCH];
#pragma unroll
    for (int kk = 0; kk < KCH; ++kk) af[kk] = frag_a(A, KD, rt * 16, kk * 32, lane);
    for (int ct = 0; ct < NT; ++ct) {
      v8f c = {};
#pragma unroll
      for (int kk = 0; kk < KCH; ++kk) {
        v16h b = frag_b(Wsw, ND, ct, kk, lane);
        c = __builtin_amdgcn_wmma_f32_16x16x32_f16(false, af[kk], false, b, (short)0, c, false, false);
      }
      int col = ct * 16 + (lane & 15);
      float s = sc[col], h0 = sh[col];
      int roff = rt * 16 + ((lane & 16) ? 8 : 0);
      float m = mx[ct];
#pragma unroll
      for (int j = 0; j < 8; ++j)
        if (vmask & (1u << (roff + j))) m = fmaxf(m, fmaf(c[j], s, h0));
      mx[ct] = m;
    }
  }
#pragma unroll
  for (int ct = 0; ct < NT; ++ct) {
    float m = fmaxf(mx[ct], __shfl_xor(mx[ct], 16, 32)); // lanes l, l^16: same column
    if (lane < 16) out[ct * 16 + lane] = (_Float16)m;
  }
}

// ---------------- embedding lookup ------------------------------------------
__global__ void embed_kernel(const int* ids, const float* emb, _Float16* out, int total) {
  int i = blockIdx.x * blockDim.x + threadIdx.x;
  if (i >= total) return;
  int n = i >> 6, c = i & 63;
  out[i] = (_Float16)emb[ids[n] * 64 + c];
}

// ---------------- weight prep: fold BN, pad, swizzle into B-fragment layout --
__global__ void prep_weights(const float* W, const float* b, const float* g, const float* beta,
                             _Float16* Wsw, float* scale, float* shift,
                             int Kin, int Nout, int KPAD, int NPAD) {
  int total = KPAD * NPAD;
  int ntiles = NPAD >> 4;
  for (int i = blockIdx.x * blockDim.x + threadIdx.x; i < total; i += gridDim.x * blockDim.x) {
    int e = i & 15;
    int lane = (i >> 4) & 31;
    int tile = i >> 9;
    int kcIdx = tile / ntiles, ct = tile % ntiles;
    int n = ct * 16 + (lane & 15);
    int k = kcIdx * 32 + ((lane & 16) ? 8 : 0) + (e < 8 ? e : 16 + (e - 8));
    float v = (k < Kin && n < Nout) ? W[(size_t)k * Nout + n] : 0.f;
    Wsw[i] = (_Float16)v;
  }
  for (int n = blockIdx.x * blockDim.x + threadIdx.x; n < NPAD; n += gridDim.x * blockDim.x) {
    float s = 0.f, h = 0.f;
    if (n < Nout) {
      if (g) { s = g[n] * BNI; h = fmaf(b[n], s, beta[n]); }
      else   { s = 1.f;        h = b[n]; }
    }
    scale[n] = s; shift[n] = h;
  }
}

// ---------------- farthest point sampling (one block per graph) --------------
template <int NPTS>
__global__ void fps_kernel(const float* pos, int* idxOut, int M) {
  __shared__ float md[NPTS];
  __shared__ float sv[256];
  __shared__ int   si[256];
  int b = blockIdx.x, t = threadIdx.x;
  const float* P = pos + (size_t)b * NPTS * 3;
  float x0 = P[0], y0 = P[1], z0 = P[2];
  for (int i = t; i < NPTS; i += 256) {
    float dx = P[3 * i] - x0, dy = P[3 * i + 1] - y0, dz = P[3 * i + 2] - z0;
    md[i] = dx * dx + dy * dy + dz * dz;
  }
  if (t == 0) idxOut[b * M] = 0;
  __syncthreads();
  for (int s = 1; s < M; ++s) {
    float best = -1.f; int bi = NPTS;
    for (int i = t; i < NPTS; i += 256) {
      float v = md[i];
      if (v > best || (v == best && i < bi)) { best = v; bi = i; }
    }
    sv[t] = best; si[t] = bi;
    __syncthreads();
    for (int off = 128; off > 0; off >>= 1) {
      if (t < off) {
        if (sv[t + off] > sv[t] || (sv[t + off] == sv[t] && si[t + off] < si[t])) {
          sv[t] = sv[t + off]; si[t] = si[t + off];
        }
      }
      __syncthreads();
    }
    int nxt = si[0];
    if (t == 0) idxOut[b * M + s] = nxt;
    float nx = P[3 * nxt], ny = P[3 * nxt + 1], nz = P[3 * nxt + 2];
    __syncthreads();
    for (int i = t; i < NPTS; i += 256) {
      float dx = P[3 * i] - nx, dy = P[3 * i + 1] - ny, dz = P[3 * i + 2] - nz;
      float d = dx * dx + dy * dy + dz * dz;
      if (d < md[i]) md[i] = d;
    }
    __syncthreads();
  }
}

// ---------------- radius ball query, K nearest within r (one wave / center) --
__global__ void ball_query(const float* pos, const int* fpsIdx, float* ctrPos, int* nbr,
                           int Npts, int M, float r2) {
  extern __shared__ char dyn_smem[];
  float* d2s = (float*)dyn_smem;
  int waveId = threadIdx.x >> 5, lane = threadIdx.x & 31;
  float* d2 = d2s + waveId * Npts;
  int center = blockIdx.x * (blockDim.x >> 5) + waveId;
  int b = center / M, m = center - b * M;
  int ci = fpsIdx[b * M + m];
  const float* P = pos + (size_t)b * Npts * 3;
  float cx = P[3 * ci], cy = P[3 * ci + 1], cz = P[3 * ci + 2];
  if (lane < 3) ctrPos[(size_t)center * 3 + lane] = P[3 * ci + lane];
  for (int i = lane; i < Npts; i += 32) {
    float dx = P[3 * i] - cx, dy = P[3 * i + 1] - cy, dz = P[3 * i + 2] - cz;
    d2[i] = dx * dx + dy * dy + dz * dz;
  }
  lds_fence_wave();
  for (int k = 0; k < 32; ++k) {
    float best = 3.0e38f; int bi = Npts;
    for (int i = lane; i < Npts; i += 32) {
      float v = d2[i];
      if (v < best || (v == best && i < bi)) { best = v; bi = i; }
    }
#pragma unroll
    for (int off = 16; off > 0; off >>= 1) {
      float ov = __shfl_xor(best, off, 32);
      int   oi = __shfl_xor(bi, off, 32);
      if (ov < best || (ov == best && oi < bi)) { best = ov; bi = oi; }
    }
    int sel = (best <= r2) ? bi : -1;
    if (lane == 0) {
      nbr[(size_t)center * 32 + k] = sel;
      if (sel >= 0) d2[sel] = 3.0e38f;
    }
    lds_fence_wave();
  }
}

// ---------------- fused SA module: gather -> 3-layer MLP (WMMA) -> max-agg ---
// Each wave processes CPW centers; weights staged once per block into LDS.
template <int CIN, int CINPAD, int C1, int C2, int COUT, int WAVES, int CPW>
__global__ void sa_fused(const _Float16* feat, const float* posIn, const float* ctrPos,
                         const int* nbr, const _Float16* gW, const float* gScales,
                         _Float16* outFeat, int Npts, int M) {
  extern __shared__ char dyn_smem[];
  constexpr int FCNT = 2 * (C1 + C2 + COUT);
  constexpr int W1SZ = CINPAD * C1, W2SZ = C1 * C2, W3SZ = C2 * COUT;
  float* fbase = (float*)dyn_smem;
  _Float16* hbase = (_Float16*)(fbase + FCNT);
  _Float16* sW1 = hbase;
  _Float16* sW2 = sW1 + W1SZ;
  _Float16* sW3 = sW2 + W2SZ;
  const float* s1 = fbase;                   const float* h1 = fbase + C1;
  const float* s2 = fbase + 2 * C1;          const float* h2 = fbase + 2 * C1 + C2;
  const float* s3 = fbase + 2 * C1 + 2 * C2; const float* h3 = fbase + 2 * C1 + 2 * C2 + COUT;

  int waveId = threadIdx.x >> 5, lane = threadIdx.x & 31;
  _Float16* wb = sW3 + W3SZ + (size_t)waveId * (32 * CINPAD + 32 * C1 + 32 * C2);
  _Float16* Aw = wb;
  _Float16* H1 = Aw + 32 * CINPAD;
  _Float16* H2 = H1 + 32 * C1;

  { // cooperative stage of weights + folded BN params into LDS
    const uint4* srcw = (const uint4*)gW;
    uint4* dstw = (uint4*)hbase;
    for (int i = threadIdx.x; i < (W1SZ + W2SZ + W3SZ) / 8; i += blockDim.x) dstw[i] = srcw[i];
    for (int i = threadIdx.x; i < FCNT; i += blockDim.x) fbase[i] = gScales[i];
  }
  __syncthreads();

  for (int it = 0; it < CPW; ++it) {
    int center = blockIdx.x * (WAVES * CPW) + waveId * CPW + it;
    int b = center / M;

    // gather: lane == neighbor row k
    int idx = nbr[(size_t)center * 32 + lane];
    unsigned vmask = (unsigned)__ballot(idx >= 0);
    bool valid = idx >= 0;
    int src = valid ? idx : 0;
    float cx = ctrPos[(size_t)center * 3 + 0];
    float cy = ctrPos[(size_t)center * 3 + 1];
    float cz = ctrPos[(size_t)center * 3 + 2];
    const _Float16* fr = feat + ((size_t)b * Npts + src) * CIN;
    const float* pr = posIn + ((size_t)b * Npts + src) * 3;
    _Float16* Arow = Aw + lane * CINPAD;
    for (int c = 0; c < CIN; ++c) Arow[c] = valid ? fr[c] : (_Float16)0.f;
    Arow[CIN + 0] = valid ? (_Float16)(pr[0] - cx) : (_Float16)0.f;
    Arow[CIN + 1] = valid ? (_Float16)(pr[1] - cy) : (_Float16)0.f;
    Arow[CIN + 2] = valid ? (_Float16)(pr[2] - cz) : (_Float16)0.f;
#pragma unroll
    for (int c = CIN + 3; c < CINPAD; ++c) Arow[c] = (_Float16)0.f;
    lds_fence_wave();

    wave_layer_t<CINPAD, C1>(Aw, sW1, s1, h1, true, H1, lane);
    lds_fence_wave();
    wave_layer_t<C1, C2>(H1, sW2, s2, h2, true, H2, lane);
    lds_fence_wave();
    wave_layer_max_t<C2, COUT>(H2, sW3, s3, h3, vmask, outFeat + (size_t)center * COUT, lane);
    lds_fence_wave();
  }
}

// ---------------- generic WMMA GEMM (one 16x16 tile per wave) ----------------
__global__ void gemm_wmma(const _Float16* A, const _Float16* Wsw,
                          const float* sc, const float* sh,
                          _Float16* outH, float* outF, int ldF, int nValid,
                          int R, int KD, int ND, int relu) {
  int wave = (blockIdx.x * blockDim.x + threadIdx.x) >> 5;
  int lane = threadIdx.x & 31;
  int tilesN = ND >> 4;
  int tiles = (R >> 4) * tilesN;
  if (wave >= tiles) return;
  int rt = wave / tilesN, ct = wave - rt * tilesN;
  v8f c = {};
  for (int kc = 0; kc < KD; kc += 32) {
    v16h a = frag_a(A, KD, rt * 16, kc, lane);
    v16h b = frag_b(Wsw, ND, ct, kc >> 5, lane);
    c = __builtin_amdgcn_wmma_f32_16x16x32_f16(false, a, false, b, (short)0, c, false, false);
  }
  int col = ct * 16 + (lane & 15);
  float s = sc[col], h0 = sh[col];
  int roff = rt * 16 + ((lane & 16) ? 8 : 0);
#pragma unroll
  for (int j = 0; j < 8; ++j) {
    float v = fmaf(c[j], s, h0);
    if (relu) v = fmaxf(v, 0.f);
    if (outH) outH[(size_t)(roff + j) * ND + col] = (_Float16)v;
    if (outF && col < nValid) outF[(size_t)(roff + j) * ldF + col] = v;
  }
}

// ---------------- SA3 input assembly and pooling -----------------------------
__global__ void build_a3(const _Float16* x2, const float* ctr2, _Float16* A3) {
  int row = blockIdx.x;
  for (int c = threadIdx.x; c < 288; c += blockDim.x) {
    _Float16 v = (_Float16)0.f;
    if (c < 256) v = x2[(size_t)row * 256 + c];
    else if (c < 259) v = (_Float16)ctr2[(size_t)row * 3 + (c - 256)];
    A3[(size_t)row * 288 + c] = v;
  }
}

__global__ void maxpool_kernel(const _Float16* h, _Float16* g, int Mrows, int C) {
  int b = blockIdx.x;
  for (int c = threadIdx.x; c < C; c += blockDim.x) {
    float mx = NEG_BIG;
    for (int m = 0; m < Mrows; ++m)
      mx = fmaxf(mx, (float)h[((size_t)b * Mrows + m) * C + c]);
    g[(size_t)b * C + c] = (_Float16)mx;
  }
}

// ============================================================================
extern "C" void kernel_launch(void* const* d_in, const int* in_sizes, int n_in,
                              void* d_out, int out_size, void* d_ws, size_t ws_size,
                              hipStream_t stream) {
  (void)in_sizes; (void)n_in; (void)out_size; (void)ws_size;
  const int*   x   = (const int*)d_in[0];
  const float* pos = (const float*)d_in[1];
  const float* emb = (const float*)d_in[2];

  char* wsp = (char*)d_ws;
  auto alloc = [&](size_t bytes) -> void* {
    void* p = (void*)wsp;
    wsp += (bytes + 255) & ~(size_t)255;
    return p;
  };

  _Float16* feat0 = (_Float16*)alloc((size_t)Bb * Nn * 64 * 2);
  int*      fps1  = (int*)alloc((size_t)Bb * M1 * 4);
  float*    ctr1  = (float*)alloc((size_t)Bb * M1 * 3 * 4);
  int*      nbr1  = (int*)alloc((size_t)Bb * M1 * 32 * 4);
  _Float16* x1    = (_Float16*)alloc((size_t)Bb * M1 * 128 * 2);
  int*      fps2  = (int*)alloc((size_t)Bb * M2 * 4);
  float*    ctr2  = (float*)alloc((size_t)Bb * M2 * 3 * 4);
  int*      nbr2  = (int*)alloc((size_t)Bb * M2 * 32 * 4);
  _Float16* x2    = (_Float16*)alloc((size_t)Bb * M2 * 256 * 2);
  _Float16* A3    = (_Float16*)alloc((size_t)Bb * M2 * 288 * 2);
  _Float16* h3a   = (_Float16*)alloc((size_t)Bb * M2 * 256 * 2);
  _Float16* h3b   = (_Float16*)alloc((size_t)Bb * M2 * 256 * 2);
  _Float16* A4    = (_Float16*)alloc((size_t)Bb * 256 * 2);
  _Float16* h4a   = (_Float16*)alloc((size_t)Bb * 256 * 2);
  _Float16* h4b   = (_Float16*)alloc((size_t)Bb * 256 * 2);

  _Float16* sa1W = (_Float16*)alloc((size_t)18432 * 2);
  float*    sa1S = (float*)alloc((size_t)512 * 4);
  _Float16* sa2W = (_Float16*)alloc((size_t)69632 * 2);
  float*    sa2S = (float*)alloc((size_t)1024 * 4);
  _Float16* W30 = (_Float16*)alloc((size_t)288 * 256 * 2); float* s30 = (float*)alloc(512 * 4);
  _Float16* W31 = (_Float16*)alloc((size_t)256 * 256 * 2); float* s31 = (float*)alloc(512 * 4);
  _Float16* W32 = (_Float16*)alloc((size_t)256 * 256 * 2); float* s32 = (float*)alloc(512 * 4);
  _Float16* W40 = (_Float16*)alloc((size_t)256 * 256 * 2); float* s40 = (float*)alloc(512 * 4);
  _Float16* W41 = (_Float16*)alloc((size_t)256 * 256 * 2); float* s41 = (float*)alloc(512 * 4);
  _Float16* W42 = (_Float16*)alloc((size_t)256 * 48 * 2);  float* s42 = (float*)alloc(96 * 4);

  auto prep = [&](int wi, int bi, int gi, int bei, int Kin, int Nout, int KPAD, int NPAD,
                  _Float16* Wsw, float* sc, float* sh) {
    int tot = KPAD * NPAD;
    prep_weights<<<(tot + 255) / 256, 256, 0, stream>>>(
        (const float*)d_in[wi], (const float*)d_in[bi],
        gi >= 0 ? (const float*)d_in[gi] : nullptr,
        bei >= 0 ? (const float*)d_in[bei] : nullptr,
        Wsw, sc, sh, Kin, Nout, KPAD, NPAD);
  };

  // sa1: 67->64 (bn), 64->64 (bn), 64->128 (plain)
  prep(3, 4, 5, 6,    67, 64,  96, 64,  sa1W,          sa1S,       sa1S + 64);
  prep(7, 8, 9, 10,   64, 64,  64, 64,  sa1W + 6144,   sa1S + 128, sa1S + 192);
  prep(11, 12, -1, -1, 64, 128, 64, 128, sa1W + 10240, sa1S + 256, sa1S + 384);
  // sa2: 131->128 (bn), 128->128 (bn), 128->256 (plain)
  prep(13, 14, 15, 16, 131, 128, 160, 128, sa2W,          sa2S,       sa2S + 128);
  prep(17, 18, 19, 20, 128, 128, 128, 128, sa2W + 20480,  sa2S + 256, sa2S + 384);
  prep(21, 22, -1, -1, 128, 256, 128, 256, sa2W + 36864,  sa2S + 512, sa2S + 768);
  // sa3: 259->256 (bn), 256->256 (bn), 256->256 (plain)
  prep(23, 24, 25, 26, 259, 256, 288, 256, W30, s30, s30 + 256);
  prep(27, 28, 29, 30, 256, 256, 256, 256, W31, s31, s31 + 256);
  prep(31, 32, -1, -1, 256, 256, 256, 256, W32, s32, s32 + 256);
  // cls: 256->256, 256->256, 256->40 (all plain; relu on hidden)
  prep(33, 34, -1, -1, 256, 256, 256, 256, W40, s40, s40 + 256);
  prep(35, 36, -1, -1, 256, 256, 256, 256, W41, s41, s41 + 256);
  prep(37, 38, -1, -1, 256, 40,  256, 48,  W42, s42, s42 + 48);

  // embedding
  embed_kernel<<<(Bb * Nn * 64 + 255) / 256, 256, 0, stream>>>(x, emb, feat0, Bb * Nn * 64);

  // ---- SA1 ----
  fps_kernel<Nn><<<Bb, 256, 0, stream>>>(pos, fps1, M1);
  ball_query<<<(Bb * M1) / 8, 256, 8 * Nn * 4, stream>>>(pos, fps1, ctr1, nbr1, Nn, M1, 0.2f * 0.2f);
  {
    size_t smem = 512 * 4 + (size_t)(18432 + 8 * (32 * 96 + 32 * 64 + 32 * 64)) * 2;
    sa_fused<64, 96, 64, 64, 128, 8, 4><<<(Bb * M1) / (8 * 4), 256, smem, stream>>>(
        feat0, pos, ctr1, nbr1, sa1W, sa1S, x1, Nn, M1);
  }

  // ---- SA2 ----
  fps_kernel<M1><<<Bb, 256, 0, stream>>>(ctr1, fps2, M2);
  ball_query<<<(Bb * M2) / 8, 256, 8 * M1 * 4, stream>>>(ctr1, fps2, ctr2, nbr2, M1, M2, 0.4f * 0.4f);
  {
    size_t smem = 1024 * 4 + (size_t)(69632 + 4 * (32 * 160 + 32 * 128 + 32 * 128)) * 2;
    sa_fused<128, 160, 128, 128, 256, 4, 4><<<(Bb * M2) / (4 * 4), 128, smem, stream>>>(
        x1, ctr1, ctr2, nbr2, sa2W, sa2S, x2, M1, M2);
  }

  // ---- SA3 (global) ----
  build_a3<<<Bb * M2, 64, 0, stream>>>(x2, ctr2, A3);
  auto gemm = [&](const _Float16* A, const _Float16* Wsw, const float* sc, const float* sh,
                  _Float16* oH, float* oF, int ldF, int nValid, int R, int KD, int ND, int relu) {
    int tiles = (R / 16) * (ND / 16);
    gemm_wmma<<<(tiles + 7) / 8, 256, 0, stream>>>(A, Wsw, sc, sh, oH, oF, ldF, nValid, R, KD, ND, relu);
  };
  gemm(A3,  W30, s30, s30 + 256, h3a, nullptr, 0, 0, Bb * M2, 288, 256, 1);
  gemm(h3a, W31, s31, s31 + 256, h3b, nullptr, 0, 0, Bb * M2, 256, 256, 1);
  gemm(h3b, W32, s32, s32 + 256, h3a, nullptr, 0, 0, Bb * M2, 256, 256, 0);
  maxpool_kernel<<<Bb, 256, 0, stream>>>(h3a, A4, M2, 256);

  // ---- classifier ----
  gemm(A4,  W40, s40, s40 + 256, h4a, nullptr, 0, 0, 16, 256, 256, 1);
  gemm(h4a, W41, s41, s41 + 256, h4b, nullptr, 0, 0, 16, 256, 256, 1);
  gemm(h4b, W42, s42, s42 + 48,  nullptr, (float*)d_out, 40, 40, 16, 256, 48, 0);
}
